// DualPtrRNNDecoder_19327352832226
// MI455X (gfx1250) — matrix operand order repaired and checked by reading, not compiled, not run on hardware
//
#include <hip/hip_runtime.h>

// ============================================================================
// DualPtrRNNDecoder on MI455X (gfx1250, wave32, WMMA bf16 16x16x32, f32 acc)
//
// Shapes: B=64, T=64, S=512, D_IN=512, H=512, 4H=2048.
//
// Plan:
//   k_f2bf / k_build_* / k_ctx / k_biassum : one-time fp32 -> bf16 staging in ws
//   k_xgemm  : X[B*T,2048] = input @ W_ih[:, :512].T + b_ih + b_hh   (WMMA GEMM)
//   k_decode : 4 workgroups x 16 batches x 1024 threads; all 64 steps in-kernel.
//              Per step: gates WMMA (N=16 = batches), LSTM pointwise in regs,
//              attention scores WMMA (h broadcast), softmax, combine WMMA
//              (attn broadcast over pre-transposed context), W_out WMMA, switch.
//
// Workspace requirement: ~112.3 MB (fits easily in 432 GiB HBM; whole working
// set is L2-resident at 192 MB).
// context_mask (jnp.bool_) is read as 1 byte per element.
// ============================================================================

typedef __attribute__((ext_vector_type(16))) __bf16 v16bf;
typedef __attribute__((ext_vector_type(8)))  float  v8f;

struct __attribute__((aligned(16))) U128 { unsigned int x, y, z, w; };

union FragU { struct { U128 lo, hi; } u; v16bf v; };
union FragS { unsigned short s[16];      v16bf v; };

__device__ __forceinline__ unsigned short f2bf(float f) {
  unsigned int b = __float_as_uint(f);
  unsigned int r = b + 0x7FFFu + ((b >> 16) & 1u);   // round-to-nearest-even
  return (unsigned short)(r >> 16);
}
__device__ __forceinline__ float bf2f(unsigned short u) {
  return __uint_as_float(((unsigned int)u) << 16);
}

// A fragment (16x32 bf16, lane = M row, same k order as B):
// lanes0-15 K=0..15, lanes16-31 K=16..31
// => one contiguous 32B chunk at byte offset kt*64 + 32*half.
__device__ __forceinline__ v16bf load_fragA(const unsigned short* row, int kt, int half) {
  FragU f;
  const char* p = (const char*)row + (size_t)kt * 64 + half * 32;
  f.u.lo = *(const U128*)(p);
  f.u.hi = *(const U128*)(p + 16);
  return f.v;
}
// B fragment (32x16 bf16, lane = N col, same k order as A: lanes0-15 K=0..15,
// lanes16-31 K=16..31) => one contiguous 32B chunk at byte offset kt*64+32*half.
__device__ __forceinline__ v16bf load_fragB(const unsigned short* row, int kt, int half) {
  FragU f;
  const char* p = (const char*)row + (size_t)kt * 64 + half * 32;
  f.u.lo = *(const U128*)(p);
  f.u.hi = *(const U128*)(p + 16);
  return f.v;
}
// B fragment converted on the fly from an f32 row (used for attention weights).
__device__ __forceinline__ v16bf load_fragB_f32(const float* row, int kt, int half) {
  FragS f;
  const float* p = row + kt * 32 + half * 16;
#pragma unroll
  for (int i = 0; i < 16; ++i) f.s[i] = f2bf(p[i]);
  return f.v;
}

__device__ __forceinline__ v8f wmma_bf16(v16bf a, v16bf b, v8f c) {
  return __builtin_amdgcn_wmma_f32_16x16x32_bf16(false, a, false, b, (short)0, c,
                                                 false, false);
}

__device__ __forceinline__ float wave_sum(float v) {
#pragma unroll
  for (int off = 16; off > 0; off >>= 1) v += __shfl_xor(v, off, 32);
  return v;
}
__device__ __forceinline__ float wave_max(float v) {
#pragma unroll
  for (int off = 16; off > 0; off >>= 1) v = fmaxf(v, __shfl_xor(v, off, 32));
  return v;
}
__device__ __forceinline__ float sigmoidf(float x) {
  return 1.0f / (1.0f + __expf(-x));
}

// ------------------------------- staging -----------------------------------
__global__ void k_f2bf(const float* __restrict__ src, unsigned short* __restrict__ dst, int n) {
  for (int i = blockIdx.x * blockDim.x + threadIdx.x; i < n; i += gridDim.x * blockDim.x)
    dst[i] = f2bf(src[i]);
}
__global__ void k_build_Wr(const float* __restrict__ Wih, const float* __restrict__ Whh,
                           unsigned short* __restrict__ Wr) {
  const int n = 2048 * 1024;
  for (int i = blockIdx.x * blockDim.x + threadIdx.x; i < n; i += gridDim.x * blockDim.x) {
    int r = i >> 10, k = i & 1023;
    float v = (k < 512) ? Wih[r * 1024 + 512 + k] : Whh[r * 512 + (k - 512)];
    Wr[i] = f2bf(v);
  }
}
__global__ void k_build_Wihin(const float* __restrict__ Wih, unsigned short* __restrict__ W) {
  const int n = 2048 * 512;
  for (int i = blockIdx.x * blockDim.x + threadIdx.x; i < n; i += gridDim.x * blockDim.x) {
    int r = i >> 9, k = i & 511;
    W[i] = f2bf(Wih[r * 1024 + k]);
  }
}
__global__ void k_ctx(const float* __restrict__ ctx, unsigned short* __restrict__ cbf,
                      unsigned short* __restrict__ cT) {
  const int n = 64 * 512 * 512;
  for (int i = blockIdx.x * blockDim.x + threadIdx.x; i < n; i += gridDim.x * blockDim.x) {
    int b = i >> 18, rem = i & 262143, s = rem >> 9, h = rem & 511;
    unsigned short v = f2bf(ctx[i]);
    cbf[i] = v;
    cT[(b << 18) + (h << 9) + s] = v;   // transposed copy for the combine GEMM
  }
}
__global__ void k_biassum(const float* __restrict__ a, const float* __restrict__ b,
                          float* __restrict__ o) {
  for (int i = blockIdx.x * blockDim.x + threadIdx.x; i < 2048; i += gridDim.x * blockDim.x)
    o[i] = a[i] + b[i];
}

// -------- X = input @ W_ih[:, :512].T + (b_ih + b_hh), M=4096 N=2048 K=512 --
__global__ void __launch_bounds__(256)
k_xgemm(const unsigned short* __restrict__ inp, const unsigned short* __restrict__ Wt,
        const float* __restrict__ bias, float* __restrict__ X) {
  const int wg   = blockIdx.x * (blockDim.x >> 5) + (threadIdx.x >> 5);
  const int lane = threadIdx.x & 31;
  const int half = lane >> 4, l16 = lane & 15;
  const int mt = wg >> 7;     // 0..255
  const int nt = wg & 127;    // 0..127
  const unsigned short* arow = inp + (size_t)(mt * 16 + l16) * 512;
  const unsigned short* brow = Wt  + (size_t)(nt * 16 + l16) * 512;
  const float bv = bias[nt * 16 + l16];
  v8f acc;
#pragma unroll
  for (int r = 0; r < 8; ++r) acc[r] = bv;
#pragma unroll
  for (int kt = 0; kt < 16; ++kt)
    acc = wmma_bf16(load_fragA(arow, kt, half), load_fragB(brow, kt, half), acc);
#pragma unroll
  for (int r = 0; r < 8; ++r)
    X[(size_t)(mt * 16 + r + 8 * half) * 2048 + nt * 16 + l16] = acc[r];
}

// ------------------------------- decoder -----------------------------------
__global__ void __launch_bounds__(1024)
k_decode(const float* __restrict__ input, const unsigned short* __restrict__ ctxbf,
         const unsigned short* __restrict__ ctxT, const unsigned char* __restrict__ mask,
         const float* __restrict__ h0, const float* __restrict__ c0,
         const unsigned short* __restrict__ Wr, const unsigned short* __restrict__ Wout,
         const float* __restrict__ Wsw, const float* __restrict__ bsw,
         const float* __restrict__ X,
         float* __restrict__ out_ctx, float* __restrict__ out_attn,
         float* __restrict__ out_align, float* __restrict__ out_sw,
         float* __restrict__ out_h, float* __restrict__ out_c) {
  // 16 batches per workgroup; 32 waves; state in LDS (bf16) + registers (f32 c)
  __shared__ __align__(16) unsigned short s_h[16][512];     // h (bf16)
  __shared__ __align__(16) unsigned short s_ctx[16][512];   // ctx_prev -> ctx_new
  __shared__ __align__(16) unsigned short s_comb[16][512];  // attention combined
  __shared__ float s_red[16][2];

  const int g = blockIdx.x;               // batch group (0..3)
  const int tid = threadIdx.x;
  const int wave = tid >> 5, lane = tid & 31;
  const int half = lane >> 4, l16 = lane & 15;

  // ---- init: ctx_out0 = 0, h = h0, c = c0 (c held in registers per wave) ----
  for (int i = tid; i < 16 * 512; i += 1024) {
    int n = i >> 9, k = i & 511;
    s_ctx[n][k] = 0;                                  // bf16(0)
    s_h[n][k]   = f2bf(h0[(size_t)(g * 16 + n) * 512 + k]);
  }
  float creg[8];
#pragma unroll
  for (int r = 0; r < 8; ++r)
    creg[r] = c0[(size_t)(g * 16 + l16) * 512 + 16 * wave + r + 8 * half];
  __syncthreads();

  for (int t = 0; t < 64; ++t) {
    float swpart = 0.0f;                              // switch-gate partial dot
    // ================= P0: gates = X + [ctx_prev; h] @ Wr.T ================
    v8f accI, accF, accG, accO;
    {
      const size_t xbase = ((size_t)((g * 16 + l16) * 64 + t)) * 2048;
      const int mrow = 16 * wave;
#pragma unroll
      for (int r = 0; r < 8; ++r) {
        int m = mrow + r + 8 * half;
        accI[r] = X[xbase + m];
        accF[r] = X[xbase + 512 + m];
        accG[r] = X[xbase + 1024 + m];
        accO[r] = X[xbase + 1536 + m];
      }
      if (t + 1 < 64) __builtin_prefetch(&X[xbase + 2048], 0, 3);  // next step
      const unsigned short* wI = Wr + (size_t)(mrow + l16) * 1024;
      const unsigned short* wF = wI + (size_t)512 * 1024;
      const unsigned short* wG = wI + (size_t)1024 * 1024;
      const unsigned short* wO = wI + (size_t)1536 * 1024;
#pragma unroll
      for (int kt = 0; kt < 32; ++kt) {
        v16bf bf = (kt < 16) ? load_fragB(&s_ctx[l16][0], kt, half)
                             : load_fragB(&s_h[l16][0], kt - 16, half);
        accI = wmma_bf16(load_fragA(wI, kt, half), bf, accI);
        accF = wmma_bf16(load_fragA(wF, kt, half), bf, accF);
        accG = wmma_bf16(load_fragA(wG, kt, half), bf, accG);
        accO = wmma_bf16(load_fragA(wO, kt, half), bf, accO);
      }
    }
    __syncthreads();   // all reads of s_ctx/s_h (prev state) complete
    // ================= P0b: LSTM pointwise, h -> LDS, c stays in regs ======
    {
#pragma unroll
      for (int r = 0; r < 8; ++r) {
        float iv = sigmoidf(accI[r]);
        float fv = sigmoidf(accF[r]);
        float gv = tanhf(accG[r]);
        float ov = sigmoidf(accO[r]);
        float c = fv * creg[r] + iv * gv;
        creg[r] = c;
        float h = ov * tanhf(c);
        int m = 16 * wave + r + 8 * half;
        s_h[l16][m] = f2bf(h);
        if (t == 63) {
          out_h[(size_t)(g * 16 + l16) * 512 + m] = h;
          out_c[(size_t)(g * 16 + l16) * 512 + m] = c;
        }
      }
    }
    __syncthreads();
    // ================= P1: scores[b,s] = context[b] @ h_new[b] =============
    // 2 waves per batch; h broadcast across the 16 B-columns; raw scores go
    // to the out_attn slab (L2-resident scratch, overwritten by softmax).
    {
      const int b = wave >> 1, sh = wave & 1;
      const unsigned short* hrow = &s_h[b][0];
      const unsigned short* cb = ctxbf + (size_t)(g * 16 + b) * 512 * 512;
      float* srow = out_attn + ((size_t)((g * 16 + b) * 64 + t)) * 512;
      for (int st = 0; st < 16; ++st) {
        const int tile = sh * 16 + st;
        const unsigned short* arow = cb + (size_t)(tile * 16 + l16) * 512;
        v8f acc;
#pragma unroll
        for (int r = 0; r < 8; ++r) acc[r] = 0.0f;
#pragma unroll
        for (int kt = 0; kt < 16; ++kt)
          acc = wmma_bf16(load_fragA(arow, kt, half), load_fragB(hrow, kt, half), acc);
        if (l16 == 0) {
#pragma unroll
          for (int r = 0; r < 8; ++r) srow[tile * 16 + r + 8 * half] = acc[r];
        }
      }
    }
    __threadfence_block();
    __syncthreads();
    // ========== P2: masked softmax (64 threads per batch) + P5a switch =====
    {
      const int n = tid >> 6, sub = tid & 63, wp = (tid >> 5) & 1;
      float* arow = out_attn + ((size_t)((g * 16 + n) * 64 + t)) * 512;
      const unsigned char* mrow = mask + (size_t)(g * 16 + n) * 512;
      float sc[8];
      float lmax = -3.402823466e38f;
#pragma unroll
      for (int q = 0; q < 8; ++q) {
        int s = sub + 64 * q;
        float v = arow[s];
        sc[q] = mrow[s] ? v : -3.402823466e38f;
        lmax = fmaxf(lmax, sc[q]);
      }
      lmax = wave_max(lmax);
      if ((tid & 31) == 0) s_red[n][wp] = lmax;
      __syncthreads();
      float mx = fmaxf(s_red[n][0], s_red[n][1]);
      float lsum = 0.0f;
#pragma unroll
      for (int q = 0; q < 8; ++q) { sc[q] = __expf(sc[q] - mx); lsum += sc[q]; }
      lsum = wave_sum(lsum);
      __syncthreads();
      if ((tid & 31) == 0) s_red[n][wp] = lsum;
      __syncthreads();
      float inv = 1.0f / (s_red[n][0] + s_red[n][1]);
#pragma unroll
      for (int q = 0; q < 8; ++q) {
        int s = sub + 64 * q;
        float a = sc[q] * inv;
        arow[s] = a;
        if (t == 63) out_align[(size_t)(g * 16 + n) * 512 + s] = a;
      }
      // P5a: switch partial over [h_new | emb_t | ctx_prev] (ctx_new later)
      const float* inrow = input + ((size_t)((g * 16 + n) * 64 + t)) * 512;
      for (int q = sub; q < 512; q += 64) {
        swpart += bf2f(s_h[n][q]) * Wsw[q];            // h_new      (q 0:512)
        swpart += inrow[q]        * Wsw[1024 + q];     // emb_t      (q 1024:1536)
        swpart += bf2f(s_ctx[n][q]) * Wsw[1536 + q];   // ctx_prev   (q 1536:2048)
      }
    }
    __threadfence_block();
    __syncthreads();
    // ======== P3: combined[b,h] = attn[b] @ context[b] (via ctxT) ==========
    {
      const int b = wave >> 1, hh = wave & 1;
      const float* arowf = out_attn + ((size_t)((g * 16 + b) * 64 + t)) * 512;
      const unsigned short* cb = ctxT + (size_t)(g * 16 + b) * 512 * 512;
      for (int ht = 0; ht < 16; ++ht) {
        const int tile = hh * 16 + ht;
        const unsigned short* arow = cb + (size_t)(tile * 16 + l16) * 512;
        v8f acc;
#pragma unroll
        for (int r = 0; r < 8; ++r) acc[r] = 0.0f;
#pragma unroll
        for (int kt = 0; kt < 16; ++kt)
          acc = wmma_bf16(load_fragA(arow, kt, half), load_fragB_f32(arowf, kt, half), acc);
        if (l16 == 0) {
#pragma unroll
          for (int r = 0; r < 8; ++r) s_comb[b][tile * 16 + r + 8 * half] = f2bf(acc[r]);
        }
      }
    }
    __syncthreads();
    // ======== P4: ctx_new = tanh(W_out @ [combined; h_new]) ================
    {
      const unsigned short* arow = Wout + (size_t)(16 * wave + l16) * 1024;
      v8f acc;
#pragma unroll
      for (int r = 0; r < 8; ++r) acc[r] = 0.0f;
#pragma unroll
      for (int kt = 0; kt < 32; ++kt) {
        v16bf bf = (kt < 16) ? load_fragB(&s_comb[l16][0], kt, half)
                             : load_fragB(&s_h[l16][0], kt - 16, half);
        acc = wmma_bf16(load_fragA(arow, kt, half), bf, acc);
      }
#pragma unroll
      for (int r = 0; r < 8; ++r) {
        float v = tanhf(acc[r]);
        int m = 16 * wave + r + 8 * half;
        s_ctx[l16][m] = f2bf(v);   // becomes ctx_prev of step t+1
        out_ctx[((size_t)((g * 16 + l16) * 64 + t)) * 512 + m] = v;
      }
    }
    __syncthreads();
    // ======== P5b: finish switch gate with ctx_new, sigmoid, store =========
    {
      const int n = tid >> 6, sub = tid & 63, wp = (tid >> 5) & 1;
      for (int q = sub; q < 512; q += 64)
        swpart += bf2f(s_ctx[n][q]) * Wsw[512 + q];    // ctx_new (q 512:1024)
      float tot = wave_sum(swpart);
      if ((tid & 31) == 0) s_red[n][wp] = tot;
      __syncthreads();
      if (sub == 0) {
        float sm = s_red[n][0] + s_red[n][1] + bsw[0];
        out_sw[(size_t)(g * 16 + n) * 64 + t] = sigmoidf(sm);
      }
    }
    __syncthreads();
  }
}

// ------------------------------- launch ------------------------------------
static constexpr size_t OFF_X     = 0;            // 4096*2048*4  = 33554432
static constexpr size_t OFF_INP   = 33554432;     // 4096*512*2   = 4194304
static constexpr size_t OFF_WR    = 37748736;     // 2048*1024*2  = 4194304
static constexpr size_t OFF_WIHIN = 41943040;     // 2048*512*2   = 2097152
static constexpr size_t OFF_WOUT  = 44040192;     // 512*1024*2   = 1048576
static constexpr size_t OFF_CTX   = 45088768;     // 64*512*512*2 = 33554432
static constexpr size_t OFF_CTXT  = 78643200;     // 33554432
static constexpr size_t OFF_BIAS  = 112197632;    // 2048*4       = 8192
// total ws requirement: 112205824 bytes

extern "C" void kernel_launch(void* const* d_in, const int* in_sizes, int n_in,
                              void* d_out, int out_size, void* d_ws, size_t ws_size,
                              hipStream_t stream) {
  (void)in_sizes; (void)n_in; (void)out_size; (void)ws_size;
  const float* input   = (const float*)d_in[0];
  const float* context = (const float*)d_in[1];
  const unsigned char* mask = (const unsigned char*)d_in[2];  // jnp.bool_ (1B)
  const float* h0   = (const float*)d_in[3];
  const float* c0   = (const float*)d_in[4];
  const float* Wih  = (const float*)d_in[5];
  const float* bih  = (const float*)d_in[6];
  const float* Whh  = (const float*)d_in[7];
  const float* bhh  = (const float*)d_in[8];
  const float* WoutF = (const float*)d_in[9];
  const float* Wsw  = (const float*)d_in[10];
  const float* bsw  = (const float*)d_in[11];

  char* ws = (char*)d_ws;
  float*          X      = (float*)(ws + OFF_X);
  unsigned short* inp_bf = (unsigned short*)(ws + OFF_INP);
  unsigned short* Wr     = (unsigned short*)(ws + OFF_WR);
  unsigned short* Wihin  = (unsigned short*)(ws + OFF_WIHIN);
  unsigned short* Wout   = (unsigned short*)(ws + OFF_WOUT);
  unsigned short* ctxbf  = (unsigned short*)(ws + OFF_CTX);
  unsigned short* ctxT   = (unsigned short*)(ws + OFF_CTXT);
  float*          bias   = (float*)(ws + OFF_BIAS);

  // one-time staging (fp32 -> bf16, transposes, fused bias)
  k_f2bf       <<<2048, 256, 0, stream>>>(input, inp_bf, 4096 * 512);
  k_build_Wr   <<<2048, 256, 0, stream>>>(Wih, Whh, Wr);
  k_build_Wihin<<<1024, 256, 0, stream>>>(Wih, Wihin);
  k_f2bf       <<< 512, 256, 0, stream>>>(WoutF, Wout, 512 * 1024);
  k_ctx        <<<8192, 256, 0, stream>>>(context, ctxbf, ctxT);
  k_biassum    <<<   8, 256, 0, stream>>>(bih, bhh, bias);

  // hoisted input-projection GEMM: X = input @ W_ih[:, :512].T + b_ih + b_hh
  k_xgemm<<<4096, 256, 0, stream>>>(inp_bf, Wihin, bias, X);

  // output slabs (flat concat in reference return order)
  float* out       = (float*)d_out;
  float* out_ctx   = out;                 // [64,64,512]
  float* out_attn  = out + 2097152;       // [64,64,512]
  float* out_align = out + 4194304;       // [64,512]
  float* out_sw    = out + 4227072;       // [64,64,1]
  float* out_h     = out + 4231168;       // [64,512]
  float* out_c     = out + 4263936;       // [64,512]

  // recurrent decode: 4 groups x 16 batches, all 64 steps inside the kernel
  k_decode<<<4, 1024, 0, stream>>>(input, ctxbf, ctxT, mask, h0, c0, Wr, Wout,
                                   Wsw, bsw, X, out_ctx, out_attn, out_align,
                                   out_sw, out_h, out_c);
}
